// GATFeatureExtractor_31482110280385
// MI455X (gfx1250) — matrix operand order repaired and checked by reading, not compiled, not run on hardware
//
#include <hip/hip_runtime.h>
#include <hip/hip_fp16.h>

#define NNODES 50000
#define NEDGES 800000
#define ETOT   (NEDGES + NNODES)   // 850000 (self loops appended)
#define FIN    128
#define HIDC   32
#define HEADS  8
#define NGRAPH 64
#define NEG_SLOPE 0.2f

typedef __attribute__((ext_vector_type(16))) _Float16 v16h;
typedef __attribute__((ext_vector_type(8)))  _Float16 v8h;
typedef __attribute__((ext_vector_type(8)))  float    v8f;

// ---------------------------------------------------------------- utilities

__global__ void k_f32_to_f16(const float* __restrict__ in,
                             _Float16* __restrict__ out, int n) {
  int i = blockIdx.x * 256 + threadIdx.x;
  if (i < n) out[i] = (_Float16)in[i];
}

__global__ void k_zero_f32(float* __restrict__ p, int n) {
  int i = blockIdx.x * 256 + threadIdx.x;
  if (i < n) p[i] = 0.0f;
}

__global__ void k_init_mz(float* __restrict__ m, float* __restrict__ z, int n) {
  int i = blockIdx.x * 256 + threadIdx.x;
  if (i < n) { m[i] = -__builtin_inff(); z[i] = 0.0f; }
}

// Pack W[K, NCOL] (f32 row-major) into per-fragment f16 layout for the
// 16x16x32 f16 WMMA B operand:
//   packed[((t*KC+kc)*32 + lane)*16 + j] = W[kc*32 + (j>>3)*16 + (lane>>4)*8 + (j&7)][t*16 + (lane&15)]
__global__ void k_packW(const float* __restrict__ W, _Float16* __restrict__ P,
                        int K, int NCOL) {
  int KC = K / 32;
  int total = (NCOL / 16) * KC * 512;
  int i = blockIdx.x * 256 + threadIdx.x;
  if (i >= total) return;
  int j   = i & 15;
  int L   = (i >> 4) & 31;
  int tkc = i >> 9;
  int kc  = tkc % KC;
  int t   = tkc / KC;
  int krow = kc * 32 + (j >> 3) * 16 + (L >> 4) * 8 + (j & 7);
  int ncol = t * 16 + (L & 15);
  P[i] = (_Float16)W[krow * NCOL + ncol];
}

// ---------------------------------------------------------------- WMMA GEMM
// C[NNODES, NCOL] = A[NNODES, KDIM] (f16 row-major) @ Bpacked  (f32 accum)
// One wave computes one 16x16 tile; K-loop of KDIM/32 WMMAs.

template <int KDIM, int NCOL>
__global__ __launch_bounds__(256) void k_gemm(const _Float16* __restrict__ A,
                                              const _Float16* __restrict__ Bp,
                                              float* __restrict__ C) {
  constexpr int NT = NCOL / 16;
  constexpr int KC = KDIM / 32;
  constexpr int MT = NNODES / 16;   // 3125, exact
  int wave = (blockIdx.x * 256 + threadIdx.x) >> 5;
  int lane = threadIdx.x & 31;
  if (wave >= MT * NT) return;      // wave-uniform: safe w.r.t. EXEC for WMMA
  int tileM = wave / NT;
  int tileN = wave % NT;
  int hi = lane >> 4, lo = lane & 15;

  // A fragment: lane<16 -> row M=lo, K = {k0..k0+7, k0+16..k0+23}
  //             lane>=16 -> row M=lo, K = {k0+8..k0+15, k0+24..k0+31}
  const _Float16* arow = A + (size_t)(tileM * 16 + lo) * KDIM + hi * 8;
  const _Float16* bptr = Bp + ((size_t)(tileN * KC) * 32 + lane) * 16;

  v8f acc = {};
#pragma unroll
  for (int kc = 0; kc < KC; ++kc) {
    v8h a0 = *(const v8h*)(arow + kc * 32);
    v8h a1 = *(const v8h*)(arow + kc * 32 + 16);
    v16h a;
#pragma unroll
    for (int j = 0; j < 8; ++j) { a[j] = a0[j]; a[j + 8] = a1[j]; }
    v16h b = *(const v16h*)(bptr + kc * 512);   // 32 lanes * 16 f16 per chunk
    acc = __builtin_amdgcn_wmma_f32_16x16x32_f16(
        false, a, false, b, (short)0, acc, false, false);
  }

  // D layout: VGPR r -> row tileM*16 + hi*8 + r, col tileN*16 + lo
  float* crow = C + (size_t)(tileM * 16 + hi * 8) * NCOL + tileN * 16 + lo;
#pragma unroll
  for (int r = 0; r < 8; ++r) crow[(size_t)r * NCOL] = acc[r];
}

// ---------------------------------------------------------------- attention

// als[n*H+h] = sum_c h[n,h,c]*a_src[h,c] ; same for ald
__global__ void k_logits(const float* __restrict__ h,
                         const float* __restrict__ asrc,
                         const float* __restrict__ adst,
                         float* __restrict__ als, float* __restrict__ ald,
                         int H) {
  int i = blockIdx.x * 256 + threadIdx.x;
  if (i >= NNODES * H) return;
  int hh = i % H;
  const float* hp = h + (size_t)i * HIDC;
  float s = 0.0f, d = 0.0f;
#pragma unroll 8
  for (int c = 0; c < HIDC; ++c) {
    float v = hp[c];
    s += v * asrc[hh * HIDC + c];
    d += v * adst[hh * HIDC + c];
  }
  als[i] = s;
  ald[i] = d;
}

__device__ __forceinline__ void atomicMaxF(float* addr, float val) {
  // signed-max / unsigned-min trick (valid for IEEE, no NaNs produced here)
  if (val >= 0.0f)
    atomicMax((int*)addr, __float_as_int(val));
  else
    atomicMin((unsigned int*)addr, (unsigned int)__float_as_int(val));
}

__device__ __forceinline__ void edge_endpoints(const int* __restrict__ ei,
                                               int e, int& s, int& d) {
  if (e < NEDGES) { s = ei[e]; d = ei[NEDGES + e]; }
  else            { s = d = e - NEDGES; }            // self loops
}

// pass A: segment max of leaky-relu logits over dst
__global__ void k_edge_max(const int* __restrict__ ei,
                           const float* __restrict__ als,
                           const float* __restrict__ ald,
                           float* __restrict__ m, int H) {
  int i = blockIdx.x * 256 + threadIdx.x;
  if (i >= ETOT * H) return;
  int e = i / H, hh = i % H;
  int s, d;
  edge_endpoints(ei, e, s, d);
  float v = als[s * H + hh] + ald[d * H + hh];
  v = v >= 0.0f ? v : NEG_SLOPE * v;
  atomicMaxF(&m[d * H + hh], v);
}

// pass B: z[dst,h] += p ; acc[dst,h,c] += p * h[src,h,c]   (one wave per edge-head)
__global__ void k_edge_acc(const int* __restrict__ ei,
                           const float* __restrict__ als,
                           const float* __restrict__ ald,
                           const float* __restrict__ m,
                           float* __restrict__ z,
                           const float* __restrict__ h,
                           float* __restrict__ acc, int H) {
  int w = (blockIdx.x * 256 + threadIdx.x) >> 5;
  int lane = threadIdx.x & 31;
  if (w >= ETOT * H) return;
  int e = w / H, hh = w % H;
  int s, d;
  edge_endpoints(ei, e, s, d);
  float v = als[s * H + hh] + ald[d * H + hh];
  v = v >= 0.0f ? v : NEG_SLOPE * v;
  float p = __expf(v - m[d * H + hh]);
  if (lane == 0) atomicAdd(&z[d * H + hh], p);
  atomicAdd(&acc[((size_t)d * H + hh) * HIDC + lane],
            p * h[((size_t)s * H + hh) * HIDC + lane]);
}

// layer1 finish: normalize, +bias, relu, emit f16 for GEMM2
__global__ void k_finish1(const float* __restrict__ acc,
                          const float* __restrict__ z,
                          const float* __restrict__ b,
                          _Float16* __restrict__ outF16) {
  int i = blockIdx.x * 256 + threadIdx.x;
  if (i >= NNODES * HEADS * HIDC) return;
  int nh = i / HIDC;                 // n*H + h
  int hc = i % (HEADS * HIDC);       // h*HIDC + c
  float v = acc[i] / z[nh] + b[hc];
  outF16[i] = (_Float16)(v > 0.0f ? v : 0.0f);
}

// layer2 finish: normalize + bias, in place
__global__ void k_finish2(float* __restrict__ acc,
                          const float* __restrict__ z,
                          const float* __restrict__ b) {
  int i = blockIdx.x * 256 + threadIdx.x;
  if (i >= NNODES * HIDC) return;
  acc[i] = acc[i] / z[i / HIDC] + b[i % HIDC];
}

// ---------------------------------------------------------------- pooling

__global__ void k_pool_acc(const float* __restrict__ h,
                           const int* __restrict__ batch,
                           float* __restrict__ sums, int* __restrict__ cnt) {
  int i = blockIdx.x * 256 + threadIdx.x;
  if (i >= NNODES * HIDC) return;
  int n = i / HIDC, c = i % HIDC;
  int g = batch[n];
  atomicAdd(&sums[g * HIDC + c], h[i]);
  if (c == 0) atomicAdd(&cnt[g], 1);
}

__global__ void k_pool_out(const float* __restrict__ sums,
                           const int* __restrict__ cnt,
                           float* __restrict__ out) {
  int i = blockIdx.x * 256 + threadIdx.x;
  if (i >= NGRAPH * HIDC) return;
  float c = (float)cnt[i / HIDC];
  out[i] = sums[i] / fmaxf(c, 1.0f);
}

// ---------------------------------------------------------------- launch

static inline size_t align256(size_t x) { return (x + 255) & ~(size_t)255; }
static inline unsigned gblk(long long n) { return (unsigned)((n + 255) / 256); }

extern "C" void kernel_launch(void* const* d_in, const int* in_sizes, int n_in,
                              void* d_out, int out_size, void* d_ws, size_t ws_size,
                              hipStream_t stream) {
  const float* x      = (const float*)d_in[0];
  const int*   ei     = (const int*)d_in[1];
  // d_in[2] edge_attr ignored (reference: lin_edge=None)
  const int*   batch  = (const int*)d_in[3];
  const float* W1     = (const float*)d_in[4];
  const float* asrc1  = (const float*)d_in[5];
  const float* adst1  = (const float*)d_in[6];
  const float* b1     = (const float*)d_in[7];
  const float* W2     = (const float*)d_in[8];
  const float* asrc2  = (const float*)d_in[9];
  const float* adst2  = (const float*)d_in[10];
  const float* b2     = (const float*)d_in[11];
  float* out = (float*)d_out;

  char* w = (char*)d_ws;
  size_t off = 0;
  auto take = [&](size_t bytes) { char* p = w + off; off = align256(off + bytes); return p; };

  _Float16* xf16 = (_Float16*)take((size_t)NNODES * FIN * 2);
  _Float16* W1p  = (_Float16*)take((size_t)FIN * HEADS * HIDC * 2);
  _Float16* W2p  = (_Float16*)take((size_t)HEADS * HIDC * HIDC * 2);
  float* h1   = (float*)take((size_t)NNODES * HEADS * HIDC * 4);
  float* als1 = (float*)take((size_t)NNODES * HEADS * 4);
  float* ald1 = (float*)take((size_t)NNODES * HEADS * 4);
  float* m1   = (float*)take((size_t)NNODES * HEADS * 4);
  float* z1   = (float*)take((size_t)NNODES * HEADS * 4);
  float* acc1 = (float*)take((size_t)NNODES * HEADS * HIDC * 4);
  _Float16* hf16 = (_Float16*)take((size_t)NNODES * HEADS * HIDC * 2);
  float* h2   = (float*)take((size_t)NNODES * HIDC * 4);
  float* als2 = (float*)take((size_t)NNODES * 4);
  float* ald2 = (float*)take((size_t)NNODES * 4);
  float* m2   = (float*)take((size_t)NNODES * 4);
  float* z2   = (float*)take((size_t)NNODES * 4);
  float* acc2 = (float*)take((size_t)NNODES * HIDC * 4);
  float* sums = (float*)take((size_t)NGRAPH * HIDC * 4 + (size_t)NGRAPH * 4);
  int*   cnt  = (int*)(sums + NGRAPH * HIDC);
  (void)ws_size; (void)n_in; (void)in_sizes; (void)out_size;

  // ---- prep
  k_f32_to_f16<<<gblk((long long)NNODES * FIN), 256, 0, stream>>>(x, xf16, NNODES * FIN);
  k_packW<<<gblk(FIN * HEADS * HIDC), 256, 0, stream>>>(W1, W1p, FIN, HEADS * HIDC);
  k_packW<<<gblk(HEADS * HIDC * HIDC), 256, 0, stream>>>(W2, W2p, HEADS * HIDC, HIDC);

  // ---- layer 1: h1 = x @ W1 (WMMA), attention, aggregation
  k_gemm<FIN, HEADS * HIDC>
      <<<(3125 * (HEADS * HIDC / 16) + 7) / 8, 256, 0, stream>>>(xf16, W1p, h1);
  k_logits<<<gblk((long long)NNODES * HEADS), 256, 0, stream>>>(h1, asrc1, adst1, als1, ald1, HEADS);
  k_init_mz<<<gblk((long long)NNODES * HEADS), 256, 0, stream>>>(m1, z1, NNODES * HEADS);
  k_zero_f32<<<gblk((long long)NNODES * HEADS * HIDC), 256, 0, stream>>>(acc1, NNODES * HEADS * HIDC);
  k_edge_max<<<gblk((long long)ETOT * HEADS), 256, 0, stream>>>(ei, als1, ald1, m1, HEADS);
  k_edge_acc<<<gblk((long long)ETOT * HEADS * 32), 256, 0, stream>>>(ei, als1, ald1, m1, z1, h1, acc1, HEADS);
  k_finish1<<<gblk((long long)NNODES * HEADS * HIDC), 256, 0, stream>>>(acc1, z1, b1, hf16);

  // ---- layer 2: h2 = relu(h1_agg) @ W2 (WMMA), attention, aggregation
  k_gemm<HEADS * HIDC, HIDC>
      <<<(3125 * (HIDC / 16) + 7) / 8, 256, 0, stream>>>(hf16, W2p, h2);
  k_logits<<<gblk((long long)NNODES), 256, 0, stream>>>(h2, asrc2, adst2, als2, ald2, 1);
  k_init_mz<<<gblk((long long)NNODES), 256, 0, stream>>>(m2, z2, NNODES);
  k_zero_f32<<<gblk((long long)NNODES * HIDC), 256, 0, stream>>>(acc2, NNODES * HIDC);
  k_edge_max<<<gblk((long long)ETOT), 256, 0, stream>>>(ei, als2, ald2, m2, 1);
  k_edge_acc<<<gblk((long long)ETOT * 32), 256, 0, stream>>>(ei, als2, ald2, m2, z2, h2, acc2, 1);
  k_finish2<<<gblk((long long)NNODES * HIDC), 256, 0, stream>>>(acc2, z2, b2);

  // ---- global mean pool
  k_zero_f32<<<gblk(NGRAPH * HIDC + NGRAPH), 256, 0, stream>>>(sums, NGRAPH * HIDC + NGRAPH);
  k_pool_acc<<<gblk((long long)NNODES * HIDC), 256, 0, stream>>>(acc2, batch, sums, cnt);
  k_pool_out<<<gblk(NGRAPH * HIDC), 256, 0, stream>>>(sums, cnt, out);
}
